// GTModel_44762149159516
// MI455X (gfx1250) — matrix operand order repaired and checked by reading, not compiled, not run on hardware
//
#include <hip/hip_runtime.h>
#include <hip/hip_bf16.h>

typedef __attribute__((ext_vector_type(16))) __bf16 v16bf;
typedef __attribute__((ext_vector_type(8)))  float  v8f;
typedef int v4i_ __attribute__((vector_size(16)));   // matches async-LDS builtin param type

#define AS1 __attribute__((address_space(1)))
#define AS3 __attribute__((address_space(3)))

#define TILE_M 128
#define TILE_N 128
#define TILE_K 32
#define NNODES 4096
#define HID 512

#if defined(__has_builtin)
#  if __has_builtin(__builtin_amdgcn_global_load_async_to_lds_b128)
#    define HAVE_ASYNC_LDS 1
#  endif
#endif
#ifndef HAVE_ASYNC_LDS
#  define HAVE_ASYNC_LDS 0
#endif

__device__ __forceinline__ void wait_async0() {
#if defined(__has_builtin) && __has_builtin(__builtin_amdgcn_s_wait_asynccnt)
    __builtin_amdgcn_s_wait_asynccnt(0);
#else
    asm volatile("s_wait_asynccnt 0x0" ::: "memory");
#endif
}

// ---------------------------------------------------------------------------
// Tiled GEMM: out = epilogue( X[MxK] @ W[KxN] + bias ), bf16 WMMA, f32 accum.
// Full-width blocks use async global->LDS DMA with double buffering (fp32
// tiles in LDS, fp32->bf16 convert at fragment build); the ragged-N tail
// block (N=20) falls back to synchronous staging with zero fill.
// mode 0: outF[M][N] fp32 (optional relu / residual / scale)
// mode 1: bf16 head-major scatter: outB[h=col&7][row][d=col>>3]   (Q, K)
// mode 2: bf16 head-major transposed: outB[h][d][row]             (V)
// ---------------------------------------------------------------------------
__global__ __launch_bounds__(256)
void gemm_bf16wmma_kernel(const float* __restrict__ X, const float* __restrict__ W,
                          const float* __restrict__ bias, const float* __restrict__ residual,
                          float* __restrict__ outF, __bf16* __restrict__ outB,
                          int M, int N, int K, float scale, int relu, int mode)
{
    __shared__ union {
        struct { float  X[2][TILE_M][TILE_K]; float  W[2][TILE_K][TILE_N]; } a;  // async path
        struct { __bf16 X[TILE_M][TILE_K];    __bf16 W[TILE_K][TILE_N];    } s;  // sync path
    } sm;

    const int t    = threadIdx.x;
    const int wave = t >> 5;
    const int lane = t & 31;
    const int hv   = lane >> 4;     // which 16-lane half
    const int l15  = lane & 15;
    const int wm   = wave >> 1;     // 0..3  -> 32-row strip
    const int wn   = wave & 1;      // 0..1  -> 64-col strip
    const int m0   = blockIdx.y * TILE_M;
    const int n0   = blockIdx.x * TILE_N;

    v8f acc[2][4];
    for (int a = 0; a < 2; ++a)
        for (int b = 0; b < 4; ++b)
            for (int e = 0; e < 8; ++e) acc[a][b][e] = 0.0f;

#if HAVE_ASYNC_LDS
    const bool fullN = (n0 + TILE_N <= N);
    if (fullN) {
        // ---- async double-buffered path -----------------------------------
        auto issue_tile = [&](int k0, int buf) {
            for (int it = 0; it < 4; ++it) {                  // X: 128x32 fp32
                int task = t + it * 256;                      // 0..1023
                int row  = task >> 3;
                int kq   = (task & 7) << 2;
                __builtin_amdgcn_global_load_async_to_lds_b128(
                    (AS1 v4i_*)(X + (size_t)(m0 + row) * K + k0 + kq),
                    (AS3 v4i_*)&sm.a.X[buf][row][kq],
                    0, 0);
            }
            for (int it = 0; it < 4; ++it) {                  // W: 32x128 fp32
                int task = t + it * 256;
                int kk   = task >> 5;
                int nn   = (task & 31) << 2;
                __builtin_amdgcn_global_load_async_to_lds_b128(
                    (AS1 v4i_*)(W + (size_t)(k0 + kk) * N + n0 + nn),
                    (AS3 v4i_*)&sm.a.W[buf][kk][nn],
                    0, 0);
            }
        };
        issue_tile(0, 0);
        const int nk = K / TILE_K;
        for (int kt = 0; kt < nk; ++kt) {
            const int buf = kt & 1;
            wait_async0();                 // my copies for this buffer landed
            __syncthreads();               // everyone's copies landed
            if (kt + 1 < nk) issue_tile((kt + 1) * TILE_K, buf ^ 1);

            v16bf af[2], bfrag[4];
            for (int rf = 0; rf < 2; ++rf) {
                int row = wm * 32 + rf * 16 + l15;
                for (int e = 0; e < 16; ++e) {
                    int kk = hv * 16 + e;   // A 16x32 layout
                    af[rf][e] = (__bf16)sm.a.X[buf][row][kk];
                }
            }
            for (int cf = 0; cf < 4; ++cf) {
                int col = wn * 64 + cf * 16 + l15;
                for (int e = 0; e < 16; ++e)                       // B 32x16 layout
                    bfrag[cf][e] = (__bf16)sm.a.W[buf][hv * 16 + e][col];
            }
            for (int rf = 0; rf < 2; ++rf)
                for (int cf = 0; cf < 4; ++cf)
                    acc[rf][cf] = __builtin_amdgcn_wmma_f32_16x16x32_bf16(
                        false, af[rf], false, bfrag[cf], (short)0, acc[rf][cf], false, false);
        }
    } else
#endif
    {
        // ---- synchronous fallback path (ragged N) -------------------------
        for (int k0 = 0; k0 < K; k0 += TILE_K) {
            for (int it = 0; it < 4; ++it) {
                int task = t + it * 256;
                int row  = task >> 3;
                int kq   = (task & 7) << 2;
                const float4 v = *(const float4*)(X + (size_t)(m0 + row) * K + k0 + kq);
                sm.s.X[row][kq + 0] = (__bf16)v.x;
                sm.s.X[row][kq + 1] = (__bf16)v.y;
                sm.s.X[row][kq + 2] = (__bf16)v.z;
                sm.s.X[row][kq + 3] = (__bf16)v.w;
            }
            for (int it = 0; it < 16; ++it) {
                int task = t + it * 256;
                int kk   = task >> 7;
                int nn   = task & 127;
                int n    = n0 + nn;
                float v  = (n < N) ? W[(size_t)(k0 + kk) * N + n] : 0.0f;
                sm.s.W[kk][nn] = (__bf16)v;
            }
            __syncthreads();
            v16bf af[2], bfrag[4];
            for (int rf = 0; rf < 2; ++rf) {
                int row = wm * 32 + rf * 16 + l15;
                for (int e = 0; e < 16; ++e) {
                    int kk = hv * 16 + e;
                    af[rf][e] = sm.s.X[row][kk];
                }
            }
            for (int cf = 0; cf < 4; ++cf) {
                int col = wn * 64 + cf * 16 + l15;
                for (int e = 0; e < 16; ++e)
                    bfrag[cf][e] = sm.s.W[hv * 16 + e][col];
            }
            for (int rf = 0; rf < 2; ++rf)
                for (int cf = 0; cf < 4; ++cf)
                    acc[rf][cf] = __builtin_amdgcn_wmma_f32_16x16x32_bf16(
                        false, af[rf], false, bfrag[cf], (short)0, acc[rf][cf], false, false);
            __syncthreads();
        }
    }

    // Epilogue (C/D layout: VGPR r -> row r (+8 for upper half), lane -> col).
    for (int rf = 0; rf < 2; ++rf)
        for (int cf = 0; cf < 4; ++cf)
            for (int r = 0; r < 8; ++r) {
                int row = m0 + wm * 32 + rf * 16 + r + hv * 8;
                int col = n0 + wn * 64 + cf * 16 + l15;
                if (col >= N) continue;
                float v = (acc[rf][cf][r] + bias[col]) * scale;
                if (relu)     v = fmaxf(v, 0.0f);
                if (residual) v += residual[(size_t)row * N + col];
                if (mode == 0) {
                    outF[(size_t)row * N + col] = v;
                } else {
                    int head = col & 7, d = col >> 3;
                    if (mode == 1) outB[((size_t)head * NNODES + row) * 64 + d] = (__bf16)v;
                    else           outB[((size_t)head * 64 + d) * NNODES + row] = (__bf16)v;
                }
            }
}

// ---------------------------------------------------------------------------
// Flash-style gated attention: out[n][d*8+h] = softmax_m(A[n][m]*(QKt)[h][n][m]) @ V
// Q,K: [8][4096][64] bf16 head-major; Vt: [8][64][4096] bf16.
// Each wave owns 16 query rows of one head; streams 32 keys per iteration.
// ---------------------------------------------------------------------------
__global__ __launch_bounds__(256)
void attn_kernel(const __bf16* __restrict__ Q, const __bf16* __restrict__ Kb,
                 const __bf16* __restrict__ Vt, const float* __restrict__ A,
                 float* __restrict__ out)
{
    __shared__ __bf16 Pl[8][16][32];   // per-wave P-tile transpose buffer

    const int t    = threadIdx.x;
    const int wave = t >> 5;
    const int lane = t & 31;
    const int hv   = lane >> 4;
    const int l15  = lane & 15;
    const int head = blockIdx.y;
    const int r0   = blockIdx.x * 128 + wave * 16;

    const __bf16* Qh = Q  + (size_t)head * NNODES * 64;
    const __bf16* Kh = Kb + (size_t)head * NNODES * 64;
    const __bf16* Vh = Vt + (size_t)head * 64 * NNODES;

    // Q fragments (16 rows x 64 dims = two K=32 A-fragments).
    v16bf qf[2];
    for (int db = 0; db < 2; ++db)
        for (int e = 0; e < 16; ++e) {
            int d = db * 32 + hv * 16 + e;
            qf[db][e] = Qh[(size_t)(r0 + l15) * 64 + d];
        }

    float mrow[8], lrow[8];
    v8f o[4];
    for (int r = 0; r < 8; ++r) { mrow[r] = -1e30f; lrow[r] = 0.0f; }
    for (int j = 0; j < 4; ++j)
        for (int e = 0; e < 8; ++e) o[j][e] = 0.0f;

    for (int c = 0; c < NNODES; c += 32) {
        __builtin_prefetch(&A[(size_t)(r0 + l15) * NNODES + c + 64], 0, 0);

        // scores: two 16x16 tiles (keys c..c+15, c+16..c+31), dim 64 in 2 steps.
        v8f s[2];
        for (int tb = 0; tb < 2; ++tb) {
            int key = c + tb * 16 + l15;
            // B-fragment (32x16): lanes0-15 dims hv*16+e of first 32, etc.
            v16bf kf0 = *(const v16bf*)(Kh + (size_t)key * 64 + hv * 16);
            v16bf kf1 = *(const v16bf*)(Kh + (size_t)key * 64 + 32 + hv * 16);
            v8f z;
            for (int e = 0; e < 8; ++e) z[e] = 0.0f;
            s[tb] = __builtin_amdgcn_wmma_f32_16x16x32_bf16(false, qf[0], false, kf0, (short)0, z,     false, false);
            s[tb] = __builtin_amdgcn_wmma_f32_16x16x32_bf16(false, qf[1], false, kf1, (short)0, s[tb], false, false);
        }

        // Gate with A (L2-resident) + online softmax update.
        float g0[8], g1[8];
        for (int r = 0; r < 8; ++r) {
            int row = r0 + r + hv * 8;
            g0[r] = A[(size_t)row * NNODES + c + l15]      * s[0][r];
            g1[r] = A[(size_t)row * NNODES + c + 16 + l15] * s[1][r];
        }
        for (int r = 0; r < 8; ++r) {
            float cm = fmaxf(g0[r], g1[r]);
            for (int off = 1; off < 16; off <<= 1) cm = fmaxf(cm, __shfl_xor(cm, off, 32));
            float mn = fmaxf(mrow[r], cm);
            float fs = __expf(mrow[r] - mn);
            float p0 = __expf(g0[r] - mn);
            float p1 = __expf(g1[r] - mn);
            float rs = p0 + p1;
            for (int off = 1; off < 16; off <<= 1) rs += __shfl_xor(rs, off, 32);
            lrow[r] = lrow[r] * fs + rs;
            mrow[r] = mn;
            for (int j = 0; j < 4; ++j) o[j][r] *= fs;
            g0[r] = p0; g1[r] = p1;
        }

        // Transpose P (C-layout -> A-fragment layout) through per-wave LDS.
        for (int r = 0; r < 8; ++r) {
            Pl[wave][r + hv * 8][l15]      = (__bf16)g0[r];
            Pl[wave][r + hv * 8][16 + l15] = (__bf16)g1[r];
        }
        __syncthreads();
        v16bf pf;
        for (int e = 0; e < 16; ++e) {
            int kk = hv * 16 + e;
            pf[e] = Pl[wave][l15][kk];
        }

        // o += P(16x32) @ V(32x64); Vt layout makes B-fragment loads contiguous.
        for (int j = 0; j < 4; ++j) {
            int dim = j * 16 + l15;
            v16bf vf = *(const v16bf*)(Vh + (size_t)dim * NNODES + c + hv * 16);
            o[j] = __builtin_amdgcn_wmma_f32_16x16x32_bf16(false, pf, false, vf, (short)0, o[j], false, false);
        }
    }

    // out[n][d*8 + h], normalized by softmax denominator.
    for (int j = 0; j < 4; ++j)
        for (int r = 0; r < 8; ++r) {
            int row = r0 + r + hv * 8;
            int d   = j * 16 + l15;
            out[(size_t)row * HID + d * 8 + head] = o[j][r] / lrow[r];
        }
}

// ---------------------------------------------------------------------------
// BatchNorm1d (training mode): per-column mean + biased var over 4096 rows.
// ---------------------------------------------------------------------------
__global__ __launch_bounds__(256)
void bn_stats_kernel(const float* __restrict__ x, int C,
                     float* __restrict__ mean, float* __restrict__ rstd)
{
    __shared__ float ssum[4][64];
    __shared__ float ssq[4][64];
    int t = threadIdx.x;
    int cidx = t & 63, seg = t >> 6;
    int col = blockIdx.x * 64 + cidx;
    float s = 0.0f, s2 = 0.0f;
    if (col < C)
        for (int r = seg; r < NNODES; r += 4) {
            float v = x[(size_t)r * C + col];
            s += v; s2 += v * v;
        }
    ssum[seg][cidx] = s; ssq[seg][cidx] = s2;
    __syncthreads();
    if (seg == 0 && col < C) {
        float S  = ssum[0][cidx] + ssum[1][cidx] + ssum[2][cidx] + ssum[3][cidx];
        float S2 = ssq[0][cidx]  + ssq[1][cidx]  + ssq[2][cidx]  + ssq[3][cidx];
        float mu  = S * (1.0f / NNODES);
        float var = S2 * (1.0f / NNODES) - mu * mu;
        mean[col] = mu;
        rstd[col] = rsqrtf(var + 1e-5f);
    }
}

__global__ void bn_apply_kernel(const float* __restrict__ x, const float* __restrict__ mean,
                                const float* __restrict__ rstd, const float* __restrict__ g,
                                const float* __restrict__ b, float* __restrict__ out,
                                int C, int total)
{
    int i = blockIdx.x * blockDim.x + threadIdx.x;
    if (i >= total) return;
    int col = i % C;
    out[i] = g[col] * (x[i] - mean[col]) * rstd[col] + b[col];
}

// ---------------------------------------------------------------------------
extern "C" void kernel_launch(void* const* d_in, const int* in_sizes, int n_in,
                              void* d_out, int out_size, void* d_ws, size_t ws_size,
                              hipStream_t stream)
{
    // jax pytree flattening order (alphabetical dict keys at each level):
    // A, features, emb.{b,w}, 3x layer{b1,b2,ffn1.{b,w},ffn2.{b,w},g1,g2,
    //   k.{b,w},o.{b,w},q.{b,w},v.{b,w}}, pred1.{b,w}, pred2.{b,w}
    const float* A      = (const float*)d_in[0];
    const float* feat   = (const float*)d_in[1];
    const float* emb_b  = (const float*)d_in[2];
    const float* emb_w  = (const float*)d_in[3];
    struct L { const float *b1,*b2,*f1b,*f1w,*f2b,*f2w,*g1,*g2,*kb,*kw,*ob,*ow,*qb,*qw,*vb,*vw; } Ls[3];
    for (int l = 0; l < 3; ++l) {
        int base = 4 + 16 * l;
        Ls[l].b1  = (const float*)d_in[base + 0];
        Ls[l].b2  = (const float*)d_in[base + 1];
        Ls[l].f1b = (const float*)d_in[base + 2];
        Ls[l].f1w = (const float*)d_in[base + 3];
        Ls[l].f2b = (const float*)d_in[base + 4];
        Ls[l].f2w = (const float*)d_in[base + 5];
        Ls[l].g1  = (const float*)d_in[base + 6];
        Ls[l].g2  = (const float*)d_in[base + 7];
        Ls[l].kb  = (const float*)d_in[base + 8];
        Ls[l].kw  = (const float*)d_in[base + 9];
        Ls[l].ob  = (const float*)d_in[base + 10];
        Ls[l].ow  = (const float*)d_in[base + 11];
        Ls[l].qb  = (const float*)d_in[base + 12];
        Ls[l].qw  = (const float*)d_in[base + 13];
        Ls[l].vb  = (const float*)d_in[base + 14];
        Ls[l].vw  = (const float*)d_in[base + 15];
    }
    const float* p1b = (const float*)d_in[52];
    const float* p1w = (const float*)d_in[53];
    const float* p2b = (const float*)d_in[54];
    const float* p2w = (const float*)d_in[55];

    char* base = (char*)d_ws;
    float*  h    = (float*)base;  base += (size_t)NNODES * 512  * sizeof(float);
    float*  tmp  = (float*)base;  base += (size_t)NNODES * 1024 * sizeof(float);
    float*  tmp2 = (float*)base;  base += (size_t)NNODES * 512  * sizeof(float);
    __bf16* qb   = (__bf16*)base; base += (size_t)8 * NNODES * 64 * sizeof(__bf16);
    __bf16* kb   = (__bf16*)base; base += (size_t)8 * NNODES * 64 * sizeof(__bf16);
    __bf16* vtb  = (__bf16*)base; base += (size_t)8 * NNODES * 64 * sizeof(__bf16);
    float*  mean = (float*)base;  base += 1024 * sizeof(float);
    float*  rstd = (float*)base;  base += 1024 * sizeof(float);

    auto gemm = [&](const float* X, const float* W, const float* bias, const float* res,
                    float* oF, __bf16* oB, int M, int N, int K, float scale, int relu, int mode) {
        dim3 grid((N + TILE_N - 1) / TILE_N, M / TILE_M);
        gemm_bf16wmma_kernel<<<grid, 256, 0, stream>>>(X, W, bias, res, oF, oB,
                                                       M, N, K, scale, relu, mode);
    };

    const float scaling = 0.044194173824159216f;  // 512^-0.5

    gemm(feat, emb_w, emb_b, nullptr, h, nullptr, NNODES, 512, 1024, 1.0f, 0, 0);
    for (int l = 0; l < 3; ++l) {
        gemm(h, Ls[l].qw, Ls[l].qb, nullptr, nullptr, qb,  NNODES, 512, 512, scaling, 0, 1);
        gemm(h, Ls[l].kw, Ls[l].kb, nullptr, nullptr, kb,  NNODES, 512, 512, 1.0f,    0, 1);
        gemm(h, Ls[l].vw, Ls[l].vb, nullptr, nullptr, vtb, NNODES, 512, 512, 1.0f,    0, 2);
        attn_kernel<<<dim3(NNODES / 128, 8), 256, 0, stream>>>(qb, kb, vtb, A, tmp);
        gemm(tmp, Ls[l].ow, Ls[l].ob, h, tmp2, nullptr, NNODES, 512, 512, 1.0f, 0, 0);
        bn_stats_kernel<<<8, 256, 0, stream>>>(tmp2, 512, mean, rstd);
        bn_apply_kernel<<<(NNODES * 512 + 255) / 256, 256, 0, stream>>>(
            tmp2, mean, rstd, Ls[l].g1, Ls[l].b1, h, 512, NNODES * 512);
        gemm(h, Ls[l].f1w, Ls[l].f1b, nullptr, tmp, nullptr, NNODES, 1024, 512, 1.0f, 1, 0);
        gemm(tmp, Ls[l].f2w, Ls[l].f2b, h, tmp2, nullptr, NNODES, 512, 1024, 1.0f, 0, 0);
        bn_stats_kernel<<<8, 256, 0, stream>>>(tmp2, 512, mean, rstd);
        bn_apply_kernel<<<(NNODES * 512 + 255) / 256, 256, 0, stream>>>(
            tmp2, mean, rstd, Ls[l].g2, Ls[l].b2, h, 512, NNODES * 512);
    }
    gemm(h, p1w, p1b, nullptr, tmp, nullptr, NNODES, 256, 512, 1.0f, 1, 0);
    gemm(tmp, p2w, p2b, nullptr, (float*)d_out, nullptr, NNODES, 20, 256, 1.0f, 0, 0);
}
